// BrainMoE_19095424598415
// MI455X (gfx1250) — compile-verified, matrix operand order checked by reading, not run on hardware
//
#include <hip/hip_runtime.h>
#include <hip/hip_bf16.h>
#include <math.h>

#define NN 30000
#define NE 480000
#define NB 60
#define DIN 200
#define KPIN 224
#define H 128
#define NEXPERT 4

typedef __attribute__((ext_vector_type(16))) _Float16 v16h;
typedef __attribute__((ext_vector_type(8)))  _Float16 v8h;
typedef __attribute__((ext_vector_type(8)))  float    v8f;

static inline int cdiv(int a, int b) { return (a + b - 1) / b; }

// ---------------------------------------------------------------------------
// WMMA GEMM, Nc fixed at 128: C[M][128] (f32) = A16[M][K] * Bt16[128][K]^T
// One wave32 per 16-row strip: 8 accumulator tiles, A fragment loaded once
// per K-step and reused for 8 WMMAs. K multiple of 32, M multiple of 16.
// Fragment layouts per CDNA5 ISA 7.12.2 (16-bit A 16x32; C 16x16 f32).
// ---------------------------------------------------------------------------
__global__ __launch_bounds__(256) void wmma_gemm128(
    const _Float16* __restrict__ A, const _Float16* __restrict__ Bt,
    float* __restrict__ C, const float* __restrict__ bias,
    _Float16* __restrict__ out16, int M, int K,
    int o16s, int accumulate, int relu)
{
  int strip = (blockIdx.x * blockDim.x + threadIdx.x) >> 5;
  int lane  = threadIdx.x & 31;
  int strips = M >> 4;
  if (strip >= strips) return;
  int row0 = strip << 4;
  int r = lane & 15, sel = lane >> 4;

  v8f acc[8];
#pragma unroll
  for (int ct = 0; ct < 8; ct++)
#pragma unroll
    for (int i = 0; i < 8; i++) acc[ct][i] = 0.0f;

  if (accumulate) {
    const float* cp = C + (size_t)(row0 + sel * 8) * 128 + r;
#pragma unroll
    for (int ct = 0; ct < 8; ct++)
#pragma unroll
      for (int i = 0; i < 8; i++) acc[ct][i] = cp[(size_t)i * 128 + ct * 16];
  }

  const _Float16* arow = A + (size_t)(row0 + r) * K + sel * 8;
  for (int k0 = 0; k0 < K; k0 += 32) {
    v8h alo = *(const v8h*)(arow + k0);
    v8h ahi = *(const v8h*)(arow + k0 + 16);
    v16h a;
#pragma unroll
    for (int i = 0; i < 8; i++) { a[i] = alo[i]; a[i + 8] = ahi[i]; }
#pragma unroll
    for (int ct = 0; ct < 8; ct++) {
      const _Float16* brow = Bt + (size_t)(ct * 16 + r) * K + sel * 8 + k0;
      v8h blo = *(const v8h*)(brow);
      v8h bhi = *(const v8h*)(brow + 16);
      v16h b;
#pragma unroll
      for (int i = 0; i < 8; i++) { b[i] = blo[i]; b[i + 8] = bhi[i]; }
      acc[ct] = __builtin_amdgcn_wmma_f32_16x16x32_f16(false, a, false, b,
                                                       (short)0, acc[ct],
                                                       false, false);
    }
  }

#pragma unroll
  for (int ct = 0; ct < 8; ct++) {
    int n = ct * 16 + r;
    float bv = bias ? bias[n] : 0.0f;
#pragma unroll
    for (int i = 0; i < 8; i++) {
      int m = row0 + i + sel * 8;
      float v = acc[ct][i] + bv;
      if (relu) v = fmaxf(v, 0.0f);
      C[(size_t)m * 128 + n] = v;
      if (out16) out16[(size_t)m * o16s + n] = (_Float16)v;
    }
  }
}

// ---------------------------------------------------------------------------
// Conversions / transposes
// ---------------------------------------------------------------------------
__global__ void cvt_pad_f16(const float* __restrict__ X, _Float16* __restrict__ Y,
                            int M, int K, int Kp)
{
  int total = M * Kp;
  for (int i = blockIdx.x * blockDim.x + threadIdx.x; i < total;
       i += gridDim.x * blockDim.x) {
    int m = i / Kp, k = i - m * Kp;
    Y[i] = (k < K) ? (_Float16)X[(size_t)m * K + k] : (_Float16)0.0f;
  }
}

__global__ void transpose_pad_f16(const float* __restrict__ W, _Float16* __restrict__ Wt,
                                  int K, int Nc, int Kp)
{
  int total = Nc * Kp;
  for (int i = blockIdx.x * blockDim.x + threadIdx.x; i < total;
       i += gridDim.x * blockDim.x) {
    int n = i / Kp, k = i - n * Kp;
    Wt[i] = (k < K) ? (_Float16)W[(size_t)k * Nc + n] : (_Float16)0.0f;
  }
}

// ---------------------------------------------------------------------------
// LayerNorm over F features (+bias pre-add, +relu), block = F threads/row
// ---------------------------------------------------------------------------
__global__ void ln_rows(const float* __restrict__ in, const float* __restrict__ bias,
                        const float* __restrict__ g, const float* __restrict__ be,
                        _Float16* __restrict__ out16, int o16s, int o16off,
                        float* __restrict__ out32, int F, int relu)
{
  __shared__ float s1[128], s2[128];
  int row = blockIdx.x, t = threadIdx.x;
  float v = in[(size_t)row * F + t] + bias[t];
  s1[t] = v; s2[t] = v * v; __syncthreads();
  for (int o = F >> 1; o > 0; o >>= 1) {
    if (t < o) { s1[t] += s1[t + o]; s2[t] += s2[t + o]; }
    __syncthreads();
  }
  float mu = s1[0] / F, var = s2[0] / F - mu * mu;
  float y = (v - mu) * rsqrtf(var + 1e-5f) * g[t] + be[t];
  if (relu) y = fmaxf(y, 0.0f);
  if (out16) out16[(size_t)row * o16s + o16off + t] = (_Float16)y;
  if (out32) out32[(size_t)row * F + t] = y;
}

// ---------------------------------------------------------------------------
// Router: softmax(logits/1.5), top-2, renormalized gates
// ---------------------------------------------------------------------------
__global__ void router_gates(const _Float16* __restrict__ hr16,
                             const float* __restrict__ rW,
                             float* __restrict__ gates, int Nn)
{
  int n = blockIdx.x * blockDim.x + threadIdx.x;
  if (n >= Nn) return;
  const _Float16* hp = hr16 + (size_t)n * 256;
  float lg[4] = {0.f, 0.f, 0.f, 0.f};
  for (int k = 0; k < 256; k++) {
    float hv = (float)hp[k];
    lg[0] += hv * rW[k * 4 + 0]; lg[1] += hv * rW[k * 4 + 1];
    lg[2] += hv * rW[k * 4 + 2]; lg[3] += hv * rW[k * 4 + 3];
  }
  const float invT = 1.0f / 1.5f;
  float mx = lg[0] * invT;
  for (int i = 1; i < 4; i++) mx = fmaxf(mx, lg[i] * invT);
  float p[4], s = 0.f;
  for (int i = 0; i < 4; i++) { p[i] = expf(lg[i] * invT - mx); s += p[i]; }
  for (int i = 0; i < 4; i++) p[i] /= s;
  int i0 = 0;
  for (int i = 1; i < 4; i++) if (p[i] > p[i0]) i0 = i;
  int i1 = -1;
  for (int i = 0; i < 4; i++)
    if (i != i0 && (i1 < 0 || p[i] > p[i1])) i1 = i;
  float s2 = fmaxf(p[i0] + p[i1], 1e-9f);
  float gv[4] = {0.f, 0.f, 0.f, 0.f};
  gv[i0] = p[i0] / s2; gv[i1] = p[i1] / s2;
  for (int i = 0; i < 4; i++) gates[(size_t)n * 4 + i] = gv[i];
}

// ---------------------------------------------------------------------------
// Degrees + normalization terms
// ---------------------------------------------------------------------------
__global__ void deg_count(const int* __restrict__ dst, float* __restrict__ deg, int E)
{
  for (int i = blockIdx.x * blockDim.x + threadIdx.x; i < E;
       i += gridDim.x * blockDim.x)
    atomicAdd(&deg[dst[i]], 1.0f);
}

__global__ void deg_inv(const float* __restrict__ deg, float* __restrict__ dinv,
                        float* __restrict__ dinvl, int n)
{
  for (int i = blockIdx.x * blockDim.x + threadIdx.x; i < n;
       i += gridDim.x * blockDim.x) {
    float d = deg[i];
    dinv[i]  = (d > 0.f) ? rsqrtf(fmaxf(d, 1.0f)) : 0.0f;
    dinvl[i] = rsqrtf(d + 1.0f);
  }
}

// out[dst] += sign * dnode[src]*dnode[dst] * x[src]   (wave per edge)
__global__ void edge_prop(const float* __restrict__ x, float* __restrict__ out,
                          const int* __restrict__ src, const int* __restrict__ dst,
                          const float* __restrict__ dnode, float sign, int E)
{
  int gw = (blockIdx.x * blockDim.x + threadIdx.x) >> 5;
  int lane = threadIdx.x & 31;
  int stride = (gridDim.x * blockDim.x) >> 5;
  for (int e = gw; e < E; e += stride) {
    int s = src[e], d = dst[e];
    float en = sign * dnode[s] * dnode[d];
    const float* xs = x + (size_t)s * H;
    float* od = out + (size_t)d * H;
#pragma unroll
    for (int j = 0; j < 4; j++) {
      int f = j * 32 + lane;
      atomicAdd(&od[f], en * xs[f]);
    }
  }
}

__global__ void self_scale_init(const float* __restrict__ X,
                                const float* __restrict__ dl,
                                float* __restrict__ Y, int total)
{
  for (int i = blockIdx.x * blockDim.x + threadIdx.x; i < total;
       i += gridDim.x * blockDim.x) {
    int n = i >> 7;
    float s = dl[n];
    Y[i] = X[i] * s * s;
  }
}

__global__ void bias_act(float* __restrict__ Y, const float* __restrict__ b,
                         _Float16* __restrict__ out16, int total, int relu)
{
  for (int i = blockIdx.x * blockDim.x + threadIdx.x; i < total;
       i += gridDim.x * blockDim.x) {
    float v = Y[i] + b[i & (H - 1)];
    if (relu) v = fmaxf(v, 0.0f);
    Y[i] = v;
    if (out16) out16[i] = (_Float16)v;
  }
}

__global__ void fill_f32(float* __restrict__ p, float v, int n)
{
  for (int i = blockIdx.x * blockDim.x + threadIdx.x; i < n;
       i += gridDim.x * blockDim.x) p[i] = v;
}

// ---------------------------------------------------------------------------
// Graph-transformer attention kernels
// ---------------------------------------------------------------------------
__device__ inline void atomicMaxF(float* addr, float val)
{
  unsigned* ua = (unsigned*)addr;
  unsigned old = *ua;
  while (true) {
    float f = __uint_as_float(old);
    if (f >= val) break;
    unsigned assumed = old;
    old = atomicCAS(ua, assumed, __float_as_uint(val));
    if (old == assumed) break;
  }
}

__global__ void edge_dot(const float* __restrict__ q, const float* __restrict__ k,
                         const int* __restrict__ src, const int* __restrict__ dst,
                         float* __restrict__ elog, float* __restrict__ m, int E)
{
  int gw = (blockIdx.x * blockDim.x + threadIdx.x) >> 5;
  int lane = threadIdx.x & 31;
  int stride = (gridDim.x * blockDim.x) >> 5;
  int h = lane >> 3, local = lane & 7;
  for (int e = gw; e < E; e += stride) {
    int s = src[e], d = dst[e];
    const float* qd = q + (size_t)d * H + h * 32 + local * 4;
    const float* ks = k + (size_t)s * H + h * 32 + local * 4;
    float sum = qd[0]*ks[0] + qd[1]*ks[1] + qd[2]*ks[2] + qd[3]*ks[3];
    sum += __shfl_xor(sum, 4);
    sum += __shfl_xor(sum, 2);
    sum += __shfl_xor(sum, 1);
    if (local == 0) {
      float v = sum * 0.17677669529663687f;  // 1/sqrt(32)
      elog[(size_t)e * 4 + h] = v;
      atomicMaxF(&m[(size_t)d * 4 + h], v);
    }
  }
}

__global__ void m_fix(float* __restrict__ m, int n)
{
  for (int i = blockIdx.x * blockDim.x + threadIdx.x; i < n;
       i += gridDim.x * blockDim.x)
    if (m[i] < -1e29f) m[i] = 0.0f;
}

__global__ void edge_softmax(float* __restrict__ elog, const float* __restrict__ m,
                             float* __restrict__ den, const int* __restrict__ dst, int E)
{
  int total = E * 4;
  for (int i = blockIdx.x * blockDim.x + threadIdx.x; i < total;
       i += gridDim.x * blockDim.x) {
    int e = i >> 2, h = i & 3;
    int d = dst[e];
    float p = expf(elog[i] - m[(size_t)d * 4 + h]);
    elog[i] = p;
    atomicAdd(&den[(size_t)d * 4 + h], p);
  }
}

__global__ void edge_agg(const float* __restrict__ elog, const float* __restrict__ den,
                         const float* __restrict__ v, const int* __restrict__ src,
                         const int* __restrict__ dst, float* __restrict__ agg, int E)
{
  int gw = (blockIdx.x * blockDim.x + threadIdx.x) >> 5;
  int lane = threadIdx.x & 31;
  int stride = (gridDim.x * blockDim.x) >> 5;
  for (int e = gw; e < E; e += stride) {
    int s = src[e], d = dst[e];
    const float* vs = v + (size_t)s * H;
    float* ad = agg + (size_t)d * H;
#pragma unroll
    for (int j = 0; j < 4; j++) {
      float attn = elog[(size_t)e * 4 + j] /
                   fmaxf(den[(size_t)d * 4 + j], 1e-9f);
      int f = j * 32 + lane;
      atomicAdd(&ad[f], attn * vs[f]);
    }
  }
}

// ---------------------------------------------------------------------------
// Expert post-norm + gated combine + residual  (block = 128 per node)
// ---------------------------------------------------------------------------
__global__ void combine_experts(const float* __restrict__ h32,
                                const float* __restrict__ e0, const float* __restrict__ e1,
                                const float* __restrict__ e2, const float* __restrict__ e3,
                                const float* __restrict__ gates,
                                const float* __restrict__ png, const float* __restrict__ pnb,
                                const float* __restrict__ escale,
                                float* __restrict__ outF)
{
  __shared__ float s1[128], s2[128];
  int n = blockIdx.x, t = threadIdx.x;
  const float* eb[4] = {e0, e1, e2, e3};
  float acc = h32[(size_t)n * H + t];
#pragma unroll
  for (int e = 0; e < 4; e++) {
    float v = eb[e][(size_t)n * H + t];
    __syncthreads();
    s1[t] = v; s2[t] = v * v; __syncthreads();
    for (int o = 64; o > 0; o >>= 1) {
      if (t < o) { s1[t] += s1[t + o]; s2[t] += s2[t + o]; }
      __syncthreads();
    }
    float mu = s1[0] / 128.0f, var = s2[0] / 128.0f - mu * mu;
    float y = (v - mu) * rsqrtf(var + 1e-5f) * png[e * H + t] + pnb[e * H + t];
    acc += gates[(size_t)n * 4 + e] * escale[e] * y;
  }
  outF[(size_t)n * H + t] = acc;
}

// ---------------------------------------------------------------------------
// Mean-pool per graph + classification head
// ---------------------------------------------------------------------------
__global__ void pool_scatter(const float* __restrict__ out, const int* __restrict__ batch,
                             float* __restrict__ pooled, float* __restrict__ counts, int Nn)
{
  int gw = (blockIdx.x * blockDim.x + threadIdx.x) >> 5;
  int lane = threadIdx.x & 31;
  int stride = (gridDim.x * blockDim.x) >> 5;
  for (int n = gw; n < Nn; n += stride) {
    int b = batch[n];
    const float* on = out + (size_t)n * H;
    float* pb = pooled + (size_t)b * H;
#pragma unroll
    for (int j = 0; j < 4; j++) {
      int f = j * 32 + lane;
      atomicAdd(&pb[f], on[f]);
    }
    if (lane == 0) atomicAdd(&counts[b], 1.0f);
  }
}

__global__ void pool_div(float* __restrict__ pooled, const float* __restrict__ counts)
{
  int b = blockIdx.x, t = threadIdx.x;
  pooled[(size_t)b * H + t] /= fmaxf(counts[b], 1.0f);
}

__global__ void head_gemm_ln(const float* __restrict__ in, int Kin,
                             const float* __restrict__ W, const float* __restrict__ b,
                             const float* __restrict__ g, const float* __restrict__ be,
                             float* __restrict__ out, int F)
{
  __shared__ float s1[128], s2[128];
  int bi = blockIdx.x, t = threadIdx.x;
  float acc = b[t];
  for (int k = 0; k < Kin; k++) acc += in[(size_t)bi * Kin + k] * W[k * F + t];
  s1[t] = acc; s2[t] = acc * acc; __syncthreads();
  for (int o = F >> 1; o > 0; o >>= 1) {
    if (t < o) { s1[t] += s1[t + o]; s2[t] += s2[t + o]; }
    __syncthreads();
  }
  float mu = s1[0] / F, var = s2[0] / F - mu * mu;
  float y = fmaxf((acc - mu) * rsqrtf(var + 1e-5f) * g[t] + be[t], 0.0f);
  out[(size_t)bi * F + t] = y;
}

__global__ void head_out(const float* __restrict__ z2, const float* __restrict__ W,
                         const float* __restrict__ b, const float* __restrict__ lb,
                         float* __restrict__ out)
{
  int bi = blockIdx.x, t = threadIdx.x;
  if (t < 2) {
    float acc = b[t] + lb[t];
    for (int k = 0; k < 64; k++) acc += z2[(size_t)bi * 64 + k] * W[k * 2 + t];
    out[(size_t)bi * 2 + t] = acc;
  }
}

// ---------------------------------------------------------------------------
// Host launcher
// ---------------------------------------------------------------------------
extern "C" void kernel_launch(void* const* d_in, const int* in_sizes, int n_in,
                              void* d_out, int out_size, void* d_ws, size_t ws_size,
                              hipStream_t stream)
{
  (void)in_sizes; (void)n_in; (void)out_size; (void)ws_size;
  const float* x      = (const float*)d_in[0];
  const float* nid    = (const float*)d_in[1];
  const int*   ei     = (const int*)d_in[2];
  const int*   batch  = (const int*)d_in[3];
  const int* src = ei, *dst = ei + NE;
  const float *fe_W=(const float*)d_in[4], *fe_b=(const float*)d_in[5],
              *fe_g=(const float*)d_in[6], *fe_be=(const float*)d_in[7];
  const float *ie_W=(const float*)d_in[8], *ie_b=(const float*)d_in[9],
              *ie_g=(const float*)d_in[10], *ie_be=(const float*)d_in[11];
  const float *fuse_W=(const float*)d_in[12], *fuse_b=(const float*)d_in[13],
              *fuse_g=(const float*)d_in[14], *fuse_be=(const float*)d_in[15];
  const float *router_W=(const float*)d_in[16];
  const float *mlp_W1=(const float*)d_in[17], *mlp_b1=(const float*)d_in[18],
              *mlp_W2=(const float*)d_in[19], *mlp_b2=(const float*)d_in[20];
  const float *cheb_W=(const float*)d_in[21], *cheb_b=(const float*)d_in[22];
  const float *gt_Wq=(const float*)d_in[23], *gt_bq=(const float*)d_in[24],
              *gt_Wk=(const float*)d_in[25], *gt_bk=(const float*)d_in[26],
              *gt_Wv=(const float*)d_in[27], *gt_bv=(const float*)d_in[28],
              *gt_Ws=(const float*)d_in[29], *gt_bs=(const float*)d_in[30];
  const float *gcn_W=(const float*)d_in[31], *gcn_b=(const float*)d_in[32];
  const float *pn_g=(const float*)d_in[33], *pn_b=(const float*)d_in[34],
              *escale=(const float*)d_in[35];
  const float *h1_W=(const float*)d_in[36], *h1_b=(const float*)d_in[37],
              *h1_g=(const float*)d_in[38], *h1_be=(const float*)d_in[39];
  const float *h2_W=(const float*)d_in[40], *h2_b=(const float*)d_in[41],
              *h2_g=(const float*)d_in[42], *h2_be=(const float*)d_in[43];
  const float *h3_W=(const float*)d_in[44], *h3_b=(const float*)d_in[45],
              *lbias=(const float*)d_in[46];
  float* out = (float*)d_out;

  // ---- workspace carve-out ----
  char* base = (char*)d_ws;
  size_t off = 0;
  auto alloc = [&](size_t bytes) -> void* {
    off = (off + 255) & ~(size_t)255;
    void* p = base + off; off += bytes; return p;
  };
  _Float16* xh16 = (_Float16*)alloc((size_t)NN * KPIN * 2);
  _Float16* id16 = (_Float16*)alloc((size_t)NN * H * 2);
  _Float16* hr16 = (_Float16*)alloc((size_t)NN * 256 * 2);
  _Float16* h16  = (_Float16*)alloc((size_t)NN * H * 2);
  _Float16* t16  = (_Float16*)alloc((size_t)NN * H * 2);
  _Float16* u16  = (_Float16*)alloc((size_t)NN * H * 2);
  _Float16* z16  = (_Float16*)alloc((size_t)NN * H * 2);
  _Float16* feWt   = (_Float16*)alloc((size_t)H * KPIN * 2);
  _Float16* ieWt   = (_Float16*)alloc((size_t)H * H * 2);
  _Float16* fuseWt = (_Float16*)alloc((size_t)H * 256 * 2);
  _Float16* mW1t   = (_Float16*)alloc((size_t)H * H * 2);
  _Float16* mW2t   = (_Float16*)alloc((size_t)H * H * 2);
  _Float16* chWt   = (_Float16*)alloc((size_t)4 * H * H * 2);
  _Float16* gqWt   = (_Float16*)alloc((size_t)2 * H * H * 2);
  _Float16* gkWt   = (_Float16*)alloc((size_t)2 * H * H * 2);
  _Float16* gvWt   = (_Float16*)alloc((size_t)2 * H * H * 2);
  _Float16* gsWt   = (_Float16*)alloc((size_t)2 * H * H * 2);
  _Float16* gcWt   = (_Float16*)alloc((size_t)2 * H * H * 2);
  float* h32  = (float*)alloc((size_t)NN * H * 4);
  float* z32  = (float*)alloc((size_t)NN * H * 4);
  float* w32  = (float*)alloc((size_t)NN * H * 4);
  float* q32  = (float*)alloc((size_t)NN * H * 4);
  float* k32  = (float*)alloc((size_t)NN * H * 4);
  float* v32  = (float*)alloc((size_t)NN * H * 4);
  float* e0   = (float*)alloc((size_t)NN * H * 4);
  float* e1   = (float*)alloc((size_t)NN * H * 4);
  float* e2   = (float*)alloc((size_t)NN * H * 4);
  float* e3   = (float*)alloc((size_t)NN * H * 4);
  float* outF = (float*)alloc((size_t)NN * H * 4);
  float* elog = (float*)alloc((size_t)NE * 4 * 4);
  float* mbuf = (float*)alloc((size_t)NN * 4 * 4);
  float* den  = (float*)alloc((size_t)NN * 4 * 4);
  float* deg  = (float*)alloc((size_t)NN * 4);
  float* dinv = (float*)alloc((size_t)NN * 4);
  float* dinvl= (float*)alloc((size_t)NN * 4);
  float* gates= (float*)alloc((size_t)NN * 4 * 4);
  float* pooled=(float*)alloc((size_t)NB * H * 4);
  float* counts=(float*)alloc((size_t)NB * 4);
  float* zh1  = (float*)alloc((size_t)NB * H * 4);
  float* zh2  = (float*)alloc((size_t)NB * 64 * 4);

  const int TB = 256;
  const dim3 EG(4096), EB(256);
  auto gemm = [&](const _Float16* A, const _Float16* Bt, float* C,
                  const float* bias, _Float16* o16, int M, int K,
                  int o16s, int acc, int relu) {
    int strips = M / 16;
    wmma_gemm128<<<cdiv(strips, 8), 256, 0, stream>>>(A, Bt, C, bias, o16,
                                                      M, K, o16s, acc, relu);
  };

  // ---- input / weight conversion ----
  cvt_pad_f16<<<cdiv(NN * KPIN, TB), TB, 0, stream>>>(x, xh16, NN, DIN, KPIN);
  cvt_pad_f16<<<cdiv(NN * H, TB), TB, 0, stream>>>(nid, id16, NN, H, H);
  transpose_pad_f16<<<cdiv(H * KPIN, TB), TB, 0, stream>>>(fe_W, feWt, DIN, H, KPIN);
  transpose_pad_f16<<<cdiv(H * H, TB), TB, 0, stream>>>(ie_W, ieWt, H, H, H);
  transpose_pad_f16<<<cdiv(H * 256, TB), TB, 0, stream>>>(fuse_W, fuseWt, 256, H, 256);
  transpose_pad_f16<<<cdiv(H * H, TB), TB, 0, stream>>>(mlp_W1, mW1t, H, H, H);
  transpose_pad_f16<<<cdiv(H * H, TB), TB, 0, stream>>>(mlp_W2, mW2t, H, H, H);
  for (int t = 0; t < 4; t++)
    transpose_pad_f16<<<cdiv(H * H, TB), TB, 0, stream>>>(cheb_W + (size_t)t * H * H,
                                                          chWt + (size_t)t * H * H, H, H, H);
  for (int l = 0; l < 2; l++) {
    transpose_pad_f16<<<cdiv(H*H,TB),TB,0,stream>>>(gt_Wq+(size_t)l*H*H, gqWt+(size_t)l*H*H, H,H,H);
    transpose_pad_f16<<<cdiv(H*H,TB),TB,0,stream>>>(gt_Wk+(size_t)l*H*H, gkWt+(size_t)l*H*H, H,H,H);
    transpose_pad_f16<<<cdiv(H*H,TB),TB,0,stream>>>(gt_Wv+(size_t)l*H*H, gvWt+(size_t)l*H*H, H,H,H);
    transpose_pad_f16<<<cdiv(H*H,TB),TB,0,stream>>>(gt_Ws+(size_t)l*H*H, gsWt+(size_t)l*H*H, H,H,H);
    transpose_pad_f16<<<cdiv(H*H,TB),TB,0,stream>>>(gcn_W+(size_t)l*H*H, gcWt+(size_t)l*H*H, H,H,H);
  }

  // ---- degrees ----
  hipMemsetAsync(deg, 0, (size_t)NN * 4, stream);
  deg_count<<<EG, EB, 0, stream>>>(dst, deg, NE);
  deg_inv<<<cdiv(NN, TB), TB, 0, stream>>>(deg, dinv, dinvl, NN);

  // ---- encoders + fuse ----
  gemm(xh16, feWt, w32, nullptr, nullptr, NN, KPIN, 0, 0, 0);
  ln_rows<<<NN, H, 0, stream>>>(w32, fe_b, fe_g, fe_be, hr16, 256, 0, nullptr, H, 1);
  gemm(id16, ieWt, w32, nullptr, nullptr, NN, H, 0, 0, 0);
  ln_rows<<<NN, H, 0, stream>>>(w32, ie_b, ie_g, ie_be, hr16, 256, 128, nullptr, H, 1);
  gemm(hr16, fuseWt, w32, nullptr, nullptr, NN, 256, 0, 0, 0);
  ln_rows<<<NN, H, 0, stream>>>(w32, fuse_b, fuse_g, fuse_be, h16, H, 0, h32, H, 1);

  // ---- router gates ----
  router_gates<<<cdiv(NN, TB), TB, 0, stream>>>(hr16, router_W, gates, NN);

  // ---- Expert 0: MLP ----
  gemm(h16, mW1t, z32, mlp_b1, t16, NN, H, H, 0, 1);
  gemm(t16, mW2t, e0, mlp_b2, nullptr, NN, H, 0, 0, 0);

  // ---- Expert 1: ChebNet K=2 ----
  hipMemsetAsync(w32, 0, (size_t)NN * H * 4, stream);
  edge_prop<<<EG, EB, 0, stream>>>(h32, w32, src, dst, dinv, -1.0f, NE);
  cvt_pad_f16<<<cdiv(NN * H, TB), TB, 0, stream>>>(w32, u16, NN, H, H);
  gemm(h16, chWt + 0 * H * H, z32, nullptr, nullptr, NN, H, 0, 0, 0);
  gemm(u16, chWt + 1 * H * H, z32, cheb_b, z16, NN, H, H, 1, 1);
  hipMemsetAsync(w32, 0, (size_t)NN * H * 4, stream);
  edge_prop<<<EG, EB, 0, stream>>>(z32, w32, src, dst, dinv, -1.0f, NE);
  cvt_pad_f16<<<cdiv(NN * H, TB), TB, 0, stream>>>(w32, u16, NN, H, H);
  gemm(z16, chWt + 2 * H * H, e1, nullptr, nullptr, NN, H, 0, 0, 0);
  gemm(u16, chWt + 3 * H * H, e1, cheb_b + H, nullptr, NN, H, 0, 1, 0);

  // ---- Expert 2: Graph Transformer ----
  // layer 0 (input h16)
  gemm(h16, gqWt, q32, gt_bq, nullptr, NN, H, 0, 0, 0);
  gemm(h16, gkWt, k32, gt_bk, nullptr, NN, H, 0, 0, 0);
  gemm(h16, gvWt, v32, gt_bv, nullptr, NN, H, 0, 0, 0);
  fill_f32<<<cdiv(NN * 4, TB), TB, 0, stream>>>(mbuf, -1e30f, NN * 4);
  hipMemsetAsync(den, 0, (size_t)NN * 4 * 4, stream);
  edge_dot<<<EG, EB, 0, stream>>>(q32, k32, src, dst, elog, mbuf, NE);
  m_fix<<<cdiv(NN * 4, TB), TB, 0, stream>>>(mbuf, NN * 4);
  edge_softmax<<<EG, EB, 0, stream>>>(elog, mbuf, den, dst, NE);
  hipMemsetAsync(w32, 0, (size_t)NN * H * 4, stream);
  edge_agg<<<EG, EB, 0, stream>>>(elog, den, v32, src, dst, w32, NE);
  gemm(h16, gsWt, w32, gt_bs, z16, NN, H, H, 1, 1);   // z = relu(agg + h@Ws + bs)
  // layer 1 (input z16)
  gemm(z16, gqWt + H * H, q32, gt_bq + H, nullptr, NN, H, 0, 0, 0);
  gemm(z16, gkWt + H * H, k32, gt_bk + H, nullptr, NN, H, 0, 0, 0);
  gemm(z16, gvWt + H * H, v32, gt_bv + H, nullptr, NN, H, 0, 0, 0);
  fill_f32<<<cdiv(NN * 4, TB), TB, 0, stream>>>(mbuf, -1e30f, NN * 4);
  hipMemsetAsync(den, 0, (size_t)NN * 4 * 4, stream);
  edge_dot<<<EG, EB, 0, stream>>>(q32, k32, src, dst, elog, mbuf, NE);
  m_fix<<<cdiv(NN * 4, TB), TB, 0, stream>>>(mbuf, NN * 4);
  edge_softmax<<<EG, EB, 0, stream>>>(elog, mbuf, den, dst, NE);
  hipMemsetAsync(e2, 0, (size_t)NN * H * 4, stream);
  edge_agg<<<EG, EB, 0, stream>>>(elog, den, v32, src, dst, e2, NE);
  gemm(z16, gsWt + H * H, e2, gt_bs + H, nullptr, NN, H, 0, 1, 0);

  // ---- Expert 3: GCN with self loops ----
  gemm(h16, gcWt, w32, nullptr, nullptr, NN, H, 0, 0, 0);
  self_scale_init<<<cdiv(NN * H, TB), TB, 0, stream>>>(w32, dinvl, z32, NN * H);
  edge_prop<<<EG, EB, 0, stream>>>(w32, z32, src, dst, dinvl, 1.0f, NE);
  bias_act<<<cdiv(NN * H, TB), TB, 0, stream>>>(z32, gcn_b, z16, NN * H, 1);
  gemm(z16, gcWt + H * H, w32, nullptr, nullptr, NN, H, 0, 0, 0);
  self_scale_init<<<cdiv(NN * H, TB), TB, 0, stream>>>(w32, dinvl, e3, NN * H);
  edge_prop<<<EG, EB, 0, stream>>>(w32, e3, src, dst, dinvl, 1.0f, NE);
  bias_act<<<cdiv(NN * H, TB), TB, 0, stream>>>(e3, gcn_b + H, nullptr, NN * H, 0);

  // ---- combine experts ----
  combine_experts<<<NN, H, 0, stream>>>(h32, e0, e1, e2, e3, gates,
                                        pn_g, pn_b, escale, outF);

  // ---- mean-pool readout ----
  hipMemsetAsync(pooled, 0, (size_t)NB * H * 4, stream);
  hipMemsetAsync(counts, 0, (size_t)NB * 4, stream);
  pool_scatter<<<EG, EB, 0, stream>>>(outF, batch, pooled, counts, NN);
  pool_div<<<NB, H, 0, stream>>>(pooled, counts);

  // ---- classification head ----
  head_gemm_ln<<<NB, H, 0, stream>>>(pooled, H, h1_W, h1_b, h1_g, h1_be, zh1, H);
  head_gemm_ln<<<NB, 64, 0, stream>>>(zh1, H, h2_W, h2_b, h2_g, h2_be, zh2, 64);
  head_out<<<NB, 32, 0, stream>>>(zh2, h3_W, h3_b, lbias, out);
}
